// SE2P_C3_20538533609539
// MI455X (gfx1250) — matrix operand order repaired and checked by reading, not compiled, not run on hardware
//
#include <hip/hip_runtime.h>
#include <hip/hip_bf16.h>
#include <cstdint>
#include <cstddef>

// ---------------------------------------------------------------------------
// Fused GNN for MI455X (gfx1250, wave32, WMMA bf16 16x16x32, f32 accum).
// One workgroup per graph; all intermediates live in registers/LDS.
// Segment-sum + global_add_pool done with ds_add_f32 in LDS.
// ---------------------------------------------------------------------------

typedef __attribute__((ext_vector_type(16))) __bf16 v16bf;
typedef __attribute__((ext_vector_type(8)))  float  v8f;

#define NUM_G   256
#define NPG     128   // nodes per graph
#define PPERT   8     // perturbations
#define HID     128
#define IN_DIM  64

// Fragment table offsets in prepped weight buffer (units: fragments of 512 bf16)
#define FRAG_L1 0     // kt=2  (K=64)
#define FRAG_L2 16    // kt=4
#define FRAG_G1 48
#define FRAG_G2 80
#define FRAG_B1 112
#define FRAG_B2 144
#define FRAG_TOTAL 176

struct U32x8 { unsigned w[8]; };   // 32 bytes = v16bf worth of packed bf16
struct U128x2 { uint4 a, b; };     // 32 bytes

__device__ __forceinline__ unsigned short f2bf_sw(float f) {
  unsigned u = __builtin_bit_cast(unsigned, f);
  u += 0x7FFFu + ((u >> 16) & 1u);           // round-to-nearest-even
  return (unsigned short)(u >> 16);
}

// pack two f32 -> two bf16 in one u32.
// Preference: clang builtin -> gfx1250 inline asm v_cvt_pk_bf16_f32 -> software.
__device__ __forceinline__ unsigned pk2(float lo, float hi) {
#if __has_builtin(__builtin_amdgcn_cvt_pk_bf16_f32)
  typedef __attribute__((ext_vector_type(2))) __bf16 v2bf;
  v2bf r = __builtin_amdgcn_cvt_pk_bf16_f32(lo, hi);
  return __builtin_bit_cast(unsigned, r);
#elif defined(__gfx1250__)
  unsigned r;
  asm("v_cvt_pk_bf16_f32 %0, %1, %2" : "=v"(r) : "v"(lo), "v"(hi));
  return r;
#else
  return (unsigned)f2bf_sw(lo) | ((unsigned)f2bf_sw(hi) << 16);
#endif
}

// A-fragment (16x32 bf16) from row-major f32 source.
// lane l: row = l%16 (baked into rowbase), K = kt*32 + (l/16)*8 + {0..7, 16..23}
__device__ __forceinline__ v16bf afrag_f32(const float* rowbase, int kt, int lane) {
  const int kA = kt * 32 + ((lane >> 4) << 3);
  const float4* p0 = (const float4*)(rowbase + kA);
  const float4* p1 = (const float4*)(rowbase + kA + 16);
  float4 a0 = p0[0], a1 = p0[1], b0 = p1[0], b1 = p1[1];
  U32x8 t;
  t.w[0] = pk2(a0.x, a0.y); t.w[1] = pk2(a0.z, a0.w);
  t.w[2] = pk2(a1.x, a1.y); t.w[3] = pk2(a1.z, a1.w);
  t.w[4] = pk2(b0.x, b0.y); t.w[5] = pk2(b0.z, b0.w);
  t.w[6] = pk2(b1.x, b1.y); t.w[7] = pk2(b1.z, b1.w);
  return __builtin_bit_cast(v16bf, t);
}

// A-fragment from a wave-private row-major bf16 LDS strip [16][HID].
__device__ __forceinline__ v16bf afrag_bf16(const unsigned short* strip, int kt, int lane) {
  const int row = lane & 15;
  const int kA  = kt * 32 + ((lane >> 4) << 3);
  const unsigned short* rp = strip + row * HID + kA;
  U128x2 t;
  t.a = *(const uint4*)rp;          // K = kA .. kA+7
  t.b = *(const uint4*)(rp + 16);   // K = kA+16 .. kA+23
  return __builtin_bit_cast(v16bf, t);
}

// B-fragment from pre-swizzled weight buffer: frag = 32 lanes x 16 bf16, lane-contiguous.
__device__ __forceinline__ v16bf bfrag(const unsigned short* wf, int frag, int lane) {
  const uint4* p = (const uint4*)(wf + ((size_t)frag * 32 + lane) * 16);
  U128x2 t; t.a = p[0]; t.b = p[1];
  return __builtin_bit_cast(v16bf, t);
}

__device__ __forceinline__ v8f wmma_bf16(v16bf a, v16bf b, v8f c) {
  return __builtin_amdgcn_wmma_f32_16x16x32_bf16(false, a, false, b, (short)0, c,
                                                 false, false);
}

// One 16xHID layer: per N-tile compute + bias + ReLU + bf16 store to LDS strip.
// Row pairs converted with one packed cvt; lo half -> row j, hi half -> row j+1.
template<int NKT>
__device__ __forceinline__ void layer_to_strip(const v16bf* a, const unsigned short* wf,
                                               int fbase, const float biasv[8],
                                               unsigned short* strip, int lane) {
  const int col = lane & 15, half = lane >> 4;
#pragma unroll
  for (int nt = 0; nt < 8; ++nt) {
    v8f c = {};
#pragma unroll
    for (int kt = 0; kt < NKT; ++kt)
      c = wmma_bf16(a[kt], bfrag(wf, fbase + kt * 8 + nt, lane), c);
    const float b = biasv[nt];
    unsigned short* p = strip + (half * 8) * HID + nt * 16 + col;
#pragma unroll
    for (int j = 0; j < 8; j += 2) {
      float v0 = fmaxf(c[j] + b, 0.f);
      float v1 = fmaxf(c[j + 1] + b, 0.f);
      unsigned pk = pk2(v0, v1);
      p[j * HID]       = (unsigned short)(pk & 0xFFFFu);   // ds_store_b16
      p[(j + 1) * HID] = (unsigned short)(pk >> 16);       // ds_store_b16_d16_hi
    }
  }
}

// ---------------------------------------------------------------------------
// Weight prep: f32 row-major [K][128] -> bf16 B-fragments.
// B layout (ISA 7.12.2): lane l holds N = nt*16 + l%16, K = kt*32 + (l/16)*8 + {0..7, 16..23}.
// ---------------------------------------------------------------------------
__global__ __launch_bounds__(256) void prep_weights(
    const float* __restrict__ Wl1, const float* __restrict__ Wl2,
    const float* __restrict__ Wg1, const float* __restrict__ Wg2,
    const float* __restrict__ Wb1, const float* __restrict__ Wb2,
    unsigned short* __restrict__ wf) {
  int tid = blockIdx.x * 256 + threadIdx.x;
  if (tid >= FRAG_TOTAL * 32) return;
  int fid = tid >> 5, lane = tid & 31;
  const float* W; int local;
  if      (fid < FRAG_L2) { W = Wl1; local = fid - FRAG_L1; }
  else if (fid < FRAG_G1) { W = Wl2; local = fid - FRAG_L2; }
  else if (fid < FRAG_G2) { W = Wg1; local = fid - FRAG_G1; }
  else if (fid < FRAG_B1) { W = Wg2; local = fid - FRAG_G2; }
  else if (fid < FRAG_B2) { W = Wb1; local = fid - FRAG_B1; }
  else                    { W = Wb2; local = fid - FRAG_B2; }
  const int kt = local >> 3, nt = local & 7;
  const int n  = nt * 16 + (lane & 15);
  const int kb = kt * 32 + ((lane >> 4) << 3);
  unsigned* dst = (unsigned*)(wf + ((size_t)fid * 32 + lane) * 16);
#pragma unroll
  for (int i = 0; i < 8; ++i)
    dst[i] = pk2(W[(size_t)(kb + 2 * i + (i >= 4 ? 8 : 0)) * HID + n],
                 W[(size_t)(kb + 2 * i + 1 + (i >= 4 ? 8 : 0)) * HID + n]);
}

// ---------------------------------------------------------------------------
// Main fused kernel: one workgroup per graph, 8 waves of 32.
// ---------------------------------------------------------------------------
__global__ __launch_bounds__(256) void fused_gnn(
    const float* __restrict__ x,
    const unsigned short* __restrict__ wf,
    const float* __restrict__ bl1, const float* __restrict__ bl2,
    const float* __restrict__ bg1, const float* __restrict__ bg2,
    const float* __restrict__ bb1, const float* __restrict__ bb2,
    const float* __restrict__ Wd1, const float* __restrict__ bd1,
    const float* __restrict__ Wd2, const float* __restrict__ bd2,
    float* __restrict__ out) {
  __shared__ float          agg[NPG][HID];          // 64 KB  (f32 for ds_add_f32)
  __shared__ unsigned short ping[8][16][HID];       // 32 KB  wave-private strips
  __shared__ unsigned short pong[8][16][HID];       // 32 KB
  __shared__ float          pooled[HID];
  __shared__ float          z1[64];
  __shared__ float          z2[16];

  const int g    = blockIdx.x;
  const int t    = threadIdx.x;
  const int wave = t >> 5, lane = t & 31;
  const int col  = lane & 15, half = lane >> 4;

  // zero shared accumulators
  float* aggf = &agg[0][0];
  for (int i = t; i < NPG * HID; i += 256) aggf[i] = 0.f;
  if (t < HID) pooled[t] = 0.f;
  __syncthreads();

  unsigned short* myping = &ping[wave][0][0];
  unsigned short* mypong = &pong[wave][0][0];
  v16bf a[4];

  // ---- Phase 1: wave = perturbation p; L1 -> L2 -> segment-sum into agg ----
  const float* xg = x + ((size_t)(g * PPERT + wave) * NPG) * IN_DIM;
  float bl1v[8], bl2v[8];
#pragma unroll
  for (int nt = 0; nt < 8; ++nt) {
    bl1v[nt] = bl1[nt * 16 + col];
    bl2v[nt] = bl2[nt * 16 + col];
  }

  for (int mt = 0; mt < 8; ++mt) {
    const float* rowb = xg + (size_t)(mt * 16 + col) * IN_DIM;
    if (mt < 7)  // prefetch next strip's rows (global_prefetch_b8)
      __builtin_prefetch(rowb + 16 * IN_DIM, 0, 3);

    a[0] = afrag_f32(rowb, 0, lane);
    a[1] = afrag_f32(rowb, 1, lane);
    layer_to_strip<2>(a, wf, FRAG_L1, bl1v, myping, lane);   // L1 (K=64)

#pragma unroll
    for (int kt = 0; kt < 4; ++kt) a[kt] = afrag_bf16(myping, kt, lane);

    // L2 per N-tile, epilogue = bias+ReLU+ds_add_f32 into agg (segment-sum)
#pragma unroll
    for (int nt = 0; nt < 8; ++nt) {
      v8f c = {};
#pragma unroll
      for (int kt = 0; kt < 4; ++kt)
        c = wmma_bf16(a[kt], bfrag(wf, FRAG_L2 + kt * 8 + nt, lane), c);
      const float b = bl2v[nt];
#pragma unroll
      for (int j = 0; j < 8; ++j) {
        float v = fmaxf(c[j] + b, 0.f);
        atomicAdd(&agg[mt * 16 + half * 8 + j][nt * 16 + col], v);
      }
    }
  }
  __syncthreads();

  // ---- Phase 2: wave = node strip; G1 -> G2 -> B1 -> B2 -> pool ----
  {
    const int mt = wave;
    float bv[8];

    const float* arow = &agg[mt * 16 + col][0];
#pragma unroll
    for (int kt = 0; kt < 4; ++kt) a[kt] = afrag_f32(arow, kt, lane);
#pragma unroll
    for (int nt = 0; nt < 8; ++nt) bv[nt] = bg1[nt * 16 + col];
    layer_to_strip<4>(a, wf, FRAG_G1, bv, myping, lane);

#pragma unroll
    for (int kt = 0; kt < 4; ++kt) a[kt] = afrag_bf16(myping, kt, lane);
#pragma unroll
    for (int nt = 0; nt < 8; ++nt) bv[nt] = bg2[nt * 16 + col];
    layer_to_strip<4>(a, wf, FRAG_G2, bv, mypong, lane);

#pragma unroll
    for (int kt = 0; kt < 4; ++kt) a[kt] = afrag_bf16(mypong, kt, lane);
#pragma unroll
    for (int nt = 0; nt < 8; ++nt) bv[nt] = bb1[nt * 16 + col];
    layer_to_strip<4>(a, wf, FRAG_B1, bv, myping, lane);

#pragma unroll
    for (int kt = 0; kt < 4; ++kt) a[kt] = afrag_bf16(myping, kt, lane);

    // B2 per N-tile: bias + ReLU + row-reduce + ds_add_f32 (global_add_pool)
#pragma unroll
    for (int nt = 0; nt < 8; ++nt) {
      v8f c = {};
#pragma unroll
      for (int kt = 0; kt < 4; ++kt)
        c = wmma_bf16(a[kt], bfrag(wf, FRAG_B2 + kt * 8 + nt, lane), c);
      const float b = bb2[nt * 16 + col];
      float s = 0.f;
#pragma unroll
      for (int j = 0; j < 8; ++j)
        s += fmaxf(c[j] + b, 0.f);
      atomicAdd(&pooled[nt * 16 + col], s);   // both lane halves add same col
    }
  }
  __syncthreads();

  // ---- Decoder (tiny): 128 -> 64 (ReLU) -> 10, log_softmax ----
  if (t < 64) {
    float s = bd1[t];
    for (int k = 0; k < HID; ++k) s += pooled[k] * Wd1[k * 64 + t];
    z1[t] = s > 0.f ? s : 0.f;
  }
  __syncthreads();
  if (t < 10) {
    float s = bd2[t];
    for (int k = 0; k < 64; ++k) s += z1[k] * Wd2[k * 10 + t];
    z2[t] = s;
  }
  __syncthreads();
  if (t < 10) {
    float m = z2[0];
    for (int i = 1; i < 10; ++i) m = fmaxf(m, z2[i]);
    float se = 0.f;
    for (int i = 0; i < 10; ++i) se += __expf(z2[i] - m);
    out[g * 10 + t] = z2[t] - m - __logf(se);
  }
}

// ---------------------------------------------------------------------------
extern "C" void kernel_launch(void* const* d_in, const int* in_sizes, int n_in,
                              void* d_out, int out_size, void* d_ws, size_t ws_size,
                              hipStream_t stream) {
  const float* x   = (const float*)d_in[0];
  // d_in[1] = ptr (int64) — structure is static, unused
  const float* Wl1 = (const float*)d_in[2];
  const float* bl1 = (const float*)d_in[3];
  const float* Wl2 = (const float*)d_in[4];
  const float* bl2 = (const float*)d_in[5];
  const float* Wg1 = (const float*)d_in[6];
  const float* bg1 = (const float*)d_in[7];
  const float* Wg2 = (const float*)d_in[8];
  const float* bg2 = (const float*)d_in[9];
  const float* Wb1 = (const float*)d_in[10];
  const float* bb1 = (const float*)d_in[11];
  const float* Wb2 = (const float*)d_in[12];
  const float* bb2 = (const float*)d_in[13];
  const float* Wd1 = (const float*)d_in[14];
  const float* bd1 = (const float*)d_in[15];
  const float* Wd2 = (const float*)d_in[16];
  const float* bd2 = (const float*)d_in[17];

  unsigned short* wf = (unsigned short*)d_ws;   // 176 KB of bf16 B-fragments
  float* out = (float*)d_out;

  prep_weights<<<(FRAG_TOTAL * 32 + 255) / 256, 256, 0, stream>>>(
      Wl1, Wl2, Wg1, Wg2, Wb1, Wb2, wf);
  fused_gnn<<<NUM_G, 256, 0, stream>>>(
      x, wf, bl1, bl2, bg1, bg2, bb1, bb2, Wd1, bd1, Wd2, bd2, out);
}
